// DeformableTransformerEncoder_37452114821809
// MI455X (gfx1250) — matrix operand order repaired and checked by reading, not compiled, not run on hardware
//
#include <hip/hip_runtime.h>
#include <math.h>

// ---------------------------------------------------------------------------
// Types for CDNA5 WMMA (wave32)
// ---------------------------------------------------------------------------
typedef __bf16 bf16_t;
typedef __attribute__((ext_vector_type(4)))  bf16_t v4bf;
typedef __attribute__((ext_vector_type(8)))  bf16_t v8bf;
typedef __attribute__((ext_vector_type(16))) bf16_t v16bf;
typedef __attribute__((ext_vector_type(8)))  float  v8f;

#define N_TOK 67200   // B * Q = 8 * 8400
#define QTOT  8400
#define DM    256

__device__ __forceinline__ float wave_sum32(float v) {
  #pragma unroll
  for (int o = 16; o; o >>= 1) v += __shfl_xor(v, o, 32);
  return v;
}

// ---------------------------------------------------------------------------
// Tiled GEMM:  out[M,N] = act( A[M,K] @ W[K,N] + bias[N] )
//   BM=128, BN=64, BK=32 ; 256 threads = 8 waves (4 along M x 2 along N),
//   each wave -> 32x32 patch = 2x2 tiles of v_wmma_f32_16x16x32_bf16.
//   Double-buffered LDS: tile k+1 global loads overlap tile k WMMAs,
//   one workgroup barrier per k-iteration.
// ACT: 0 = none, 1 = exact GELU.  A_BF16: A operand already bf16 in global.
// All offsets 32-bit so codegen can use SGPR-base + voffset addressing.
// ---------------------------------------------------------------------------
template <int ACT, bool A_BF16, bool OUT_BF16>
__global__ __launch_bounds__(256) void gemm_bias_kernel(
    const void* __restrict__ Ap, const float* __restrict__ W,
    const float* __restrict__ bias, void* __restrict__ Op,
    int K, int Nout) {
  constexpr int BM = 128, BN = 64, BK = 32;
  __shared__ __attribute__((aligned(128))) bf16_t As[2][BM][BK];
  __shared__ __attribute__((aligned(128))) bf16_t Bs[2][BN][BK];  // [n][k]

  const int tid  = threadIdx.x;
  const int lane = tid & 31;
  const int wid  = tid >> 5;
  const int wm   = (wid & 3) * 32;   // wave row offset in block tile
  const int wn   = (wid >> 2) * 32;  // wave col offset in block tile
  const unsigned m0 = blockIdx.x * BM;
  const unsigned n0 = blockIdx.y * BN;
  const unsigned uK = (unsigned)K, uN = (unsigned)Nout;
  const bool fullN = (int)(n0 + BN) <= Nout;   // partial tile only for Nout=96

  // register staging for the in-flight tile
  float4 aR[4];   // A, f32 source path
  v8bf   aRb[2];  // A, bf16 source path
  float4 bR[2];   // B, vectorized path (full N-tile)
  float  bS[8];   // B, guarded scalar path (partial N-tile)

  auto load_tile = [&](unsigned k0) {
    if constexpr (A_BF16) {
      const bf16_t* A = (const bf16_t*)Ap;
      #pragma unroll
      for (int i = 0; i < 2; ++i) {
        unsigned idx = tid + 256u * i;           // 512 chunks of 8 bf16
        unsigned row = idx >> 2, kc = (idx & 3) * 8;
        aRb[i] = *(const v8bf*)&A[(m0 + row) * uK + k0 + kc];
      }
    } else {
      const float* A = (const float*)Ap;
      #pragma unroll
      for (int i = 0; i < 4; ++i) {
        unsigned idx = tid + 256u * i;           // 1024 chunks of float4
        unsigned row = idx >> 3, kc = (idx & 7) * 4;
        aR[i] = *(const float4*)&A[(m0 + row) * uK + k0 + kc];
      }
    }
    if (fullN) {
      #pragma unroll
      for (int i = 0; i < 2; ++i) {
        unsigned c = tid * 2u + i;               // 512 float4 chunks, N-major
        unsigned k = c >> 4, nc = (c & 15) * 4;
        bR[i] = *(const float4*)&W[(k0 + k) * uN + n0 + nc];
      }
    } else {
      #pragma unroll
      for (int e = 0; e < 8; ++e) {
        unsigned lin = tid * 8u + e;
        unsigned n = lin >> 5, k = lin & 31;
        int col = (int)(n0 + n);
        bS[e] = (col < Nout) ? W[(k0 + k) * uN + (unsigned)col] : 0.0f;
      }
    }
  };

  auto store_tile = [&](int buf) {
    if constexpr (A_BF16) {
      #pragma unroll
      for (int i = 0; i < 2; ++i) {
        unsigned idx = tid + 256u * i;
        unsigned row = idx >> 2, kc = (idx & 3) * 8;
        *(v8bf*)&As[buf][row][kc] = aRb[i];
      }
    } else {
      #pragma unroll
      for (int i = 0; i < 4; ++i) {
        unsigned idx = tid + 256u * i;
        unsigned row = idx >> 3, kc = (idx & 7) * 4;
        v4bf t;
        t[0] = (bf16_t)aR[i].x; t[1] = (bf16_t)aR[i].y;
        t[2] = (bf16_t)aR[i].z; t[3] = (bf16_t)aR[i].w;
        *(v4bf*)&As[buf][row][kc] = t;
      }
    }
    if (fullN) {
      #pragma unroll
      for (int i = 0; i < 2; ++i) {
        unsigned c = tid * 2u + i;
        unsigned k = c >> 4, nc = (c & 15) * 4;
        const float fv[4] = {bR[i].x, bR[i].y, bR[i].z, bR[i].w};
        #pragma unroll
        for (int j = 0; j < 4; ++j) Bs[buf][nc + j][k] = (bf16_t)fv[j];
      }
    } else {
      #pragma unroll
      for (int e = 0; e < 8; ++e) {
        unsigned lin = tid * 8u + e;
        Bs[buf][lin >> 5][lin & 31] = (bf16_t)bS[e];
      }
    }
  };

  v8f acc[2][2] = {};
  const int nkt = K / BK;

  load_tile(0);
  store_tile(0);
  __syncthreads();

  const int mrow = lane & 15;
  const int half = lane >> 4;

  for (int kt = 0; kt < nkt; ++kt) {
    const int cur = kt & 1;
    const bool more = (kt + 1 < nkt);
    if (more) load_tile((unsigned)(kt + 1) * BK);  // in flight during WMMAs

    // ---- fragments per ISA 16-bit layouts (two b128 LDS reads each) ----
    v16bf a[2], b[2];
    #pragma unroll
    for (int t = 0; t < 2; ++t) {
      // A 16x32: lane half selects K blocks {0..7,16..23} vs {8..15,24..31}
      const bf16_t* ap = &As[cur][wm + t * 16 + mrow][half * 8];
      v8bf alo = *(const v8bf*)ap;
      v8bf ahi = *(const v8bf*)(ap + 16);
      a[t] = __builtin_shufflevector(alo, ahi, 0, 1, 2, 3, 4, 5, 6, 7,
                                     8, 9, 10, 11, 12, 13, 14, 15);
      // B 32x16: lane = column, lane half selects K blocks {0..7,16..23} vs {8..15,24..31}
      const bf16_t* bp = &Bs[cur][wn + t * 16 + mrow][half * 8];
      v8bf blo = *(const v8bf*)bp;
      v8bf bhi = *(const v8bf*)(bp + 16);
      b[t] = __builtin_shufflevector(blo, bhi, 0, 1, 2, 3, 4, 5, 6, 7,
                                     8, 9, 10, 11, 12, 13, 14, 15);
    }

    #pragma unroll
    for (int tm = 0; tm < 2; ++tm)
      #pragma unroll
      for (int tn = 0; tn < 2; ++tn)
        acc[tm][tn] = __builtin_amdgcn_wmma_f32_16x16x32_bf16(
            false, a[tm], false, b[tn], (short)0, acc[tm][tn], false, false);

    if (more) store_tile(cur ^ 1);
    __syncthreads();
  }

  // ---- epilogue: C/D layout -> lane<16: N=lane, M=r ; lane>=16: M=8+r ----
  #pragma unroll
  for (int tm = 0; tm < 2; ++tm) {
    const unsigned rowb = m0 + wm + tm * 16 + (lane >> 4) * 8;
    #pragma unroll
    for (int tn = 0; tn < 2; ++tn) {
      const int col = (int)(n0 + wn + tn * 16 + (lane & 15));
      if (col >= Nout) continue;
      const float bv = bias[col];
      #pragma unroll
      for (int r = 0; r < 8; ++r) {
        float v = acc[tm][tn][r] + bv;
        if constexpr (ACT == 1)
          v = 0.5f * v * (1.0f + erff(v * 0.70710678118654752f));
        if constexpr (OUT_BF16)
          ((bf16_t*)Op)[(rowb + r) * uN + (unsigned)col] = (bf16_t)v;
        else
          ((float*)Op)[(rowb + r) * uN + (unsigned)col] = v;
      }
    }
  }
}

// ---------------------------------------------------------------------------
// Multi-scale deformable attention sampling.
// One wave per (b, q, head); lane = channel d in [0,32).
// value: [B,Q,8,32] f32 ; offs: [B,Q,192] ; attn logits: [B,Q,96]
// out: [B,Q,256]
// ---------------------------------------------------------------------------
__global__ __launch_bounds__(256) void deform_attn_kernel(
    const float* __restrict__ value, const float* __restrict__ offs,
    const float* __restrict__ attn, float* __restrict__ out) {
  const int lane = threadIdx.x & 31;
  const unsigned gid  = blockIdx.x * 8u + (threadIdx.x >> 5);  // (b*Q+q)*8+head
  const unsigned head = gid & 7;
  const unsigned bq   = gid >> 3;
  const unsigned q    = bq % QTOT;
  const unsigned b    = bq / QTOT;

  // query's own reference point (pixel center of its level grid)
  unsigned ls, lw;
  if (q < 6400u)      { ls = 0;    lw = 80; }
  else if (q < 8000u) { ls = 6400; lw = 40; }
  else                { ls = 8000; lw = 20; }
  const unsigned local = q - ls;
  const float rx = ((local % lw) + 0.5f) / (float)lw;
  const float ry = ((local / lw) + 0.5f) / (float)lw;  // H == W per level

  // softmax over 12 attention logits (redundant across lanes; L0-resident)
  const float* al = attn + bq * 96u + head * 12u;
  float w[12];
  float mx = -1e30f;
  #pragma unroll
  for (int i = 0; i < 12; ++i) { w[i] = al[i]; mx = fmaxf(mx, w[i]); }
  float s = 0.f;
  #pragma unroll
  for (int i = 0; i < 12; ++i) { w[i] = __expf(w[i] - mx); s += w[i]; }
  const float inv = 1.0f / s;

  const float* of    = offs + bq * 192u + head * 24u;
  const float* vbase = value + b * (unsigned)(QTOT * DM) + head * 32u + lane;

  const int LS[3] = {0, 6400, 8000};
  const int LW[3] = {80, 40, 20};

  float acc = 0.f;
  #pragma unroll
  for (int l = 0; l < 3; ++l) {
    const int Wl = LW[l], Hl = LW[l], st = LS[l];
    const float fW = (float)Wl, fH = (float)Hl;
    #pragma unroll
    for (int p = 0; p < 4; ++p) {
      const float ox = of[(l * 4 + p) * 2 + 0];
      const float oy = of[(l * 4 + p) * 2 + 1];
      const float x  = (rx + ox / fW) * fW - 0.5f;
      const float y  = (ry + oy / fH) * fH - 0.5f;
      const float x0f = floorf(x), y0f = floorf(y);
      const float fx = x - x0f, fy = y - y0f;
      const int x0 = (int)x0f, y0 = (int)y0f;
      float sample = 0.f;
      #pragma unroll
      for (int c = 0; c < 4; ++c) {
        const int xi = x0 + (c & 1);
        const int yi = y0 + (c >> 1);
        const float wt = ((c & 1) ? fx : 1.0f - fx) * ((c >> 1) ? fy : 1.0f - fy);
        if (xi >= 0 && xi < Wl && yi >= 0 && yi < Hl)
          sample += wt * vbase[(unsigned)(st + yi * Wl + xi) * (unsigned)DM];
      }
      acc += (w[l * 4 + p] * inv) * sample;
    }
  }
  out[bq * (unsigned)DM + head * 32u + lane] = acc;
}

// ---------------------------------------------------------------------------
// y = LayerNorm(x + r) * g + be ; one wave per token (8 values / lane)
// ---------------------------------------------------------------------------
__global__ __launch_bounds__(256) void add_ln_kernel(
    const float* __restrict__ x, const float* __restrict__ r,
    const float* __restrict__ g, const float* __restrict__ be,
    float* __restrict__ out) {
  const int lane = threadIdx.x & 31;
  const unsigned tok = blockIdx.x * 8u + (threadIdx.x >> 5);
  const float* xr = x + tok * (unsigned)DM;
  const float* rr = r + tok * (unsigned)DM;

  float v[8];
  float s = 0.f;
  #pragma unroll
  for (int i = 0; i < 8; ++i) {
    v[i] = xr[lane + 32 * i] + rr[lane + 32 * i];
    s += v[i];
  }
  s = wave_sum32(s);
  const float mu = s * (1.0f / 256.0f);
  float vs = 0.f;
  #pragma unroll
  for (int i = 0; i < 8; ++i) { const float d = v[i] - mu; vs += d * d; }
  vs = wave_sum32(vs);
  const float rstd = rsqrtf(vs * (1.0f / 256.0f) + 1e-5f);

  float* orow = out + tok * (unsigned)DM;
  #pragma unroll
  for (int i = 0; i < 8; ++i) {
    const int col = lane + 32 * i;
    orow[col] = (v[i] - mu) * rstd * g[col] + be[col];
  }
}

// ---------------------------------------------------------------------------
// Host orchestration
// ---------------------------------------------------------------------------
extern "C" void kernel_launch(void* const* d_in, const int* in_sizes, int n_in,
                              void* d_out, int out_size, void* d_ws,
                              size_t ws_size, hipStream_t stream) {
  (void)in_sizes; (void)n_in; (void)out_size; (void)ws_size;
  const float* feat = (const float*)d_in[0];
  auto P = [&](int layer, int j) -> const float* {
    return (const float*)d_in[1 + layer * 16 + j];
  };
  // param order: 0 W_off 1 b_off 2 W_attn 3 b_attn 4 W_val 5 b_val
  //              6 W_out 7 b_out 8 W1 9 b1 10 W2 11 b2 12 g1 13 be1 14 g2 15 be2

  const size_t N = N_TOK;
  char* ws = (char*)d_ws;
  float*  buf_x   = (float*)ws;  ws += N * DM * sizeof(float);   // current feat
  float*  buf_y   = (float*)ws;  ws += N * DM * sizeof(float);   // post-attn LN
  float*  buf_v   = (float*)ws;  ws += N * DM * sizeof(float);   // value / src2
  float*  buf_ao  = (float*)ws;  ws += N * DM * sizeof(float);   // attn out / ffn out
  float*  buf_off = (float*)ws;  ws += N * 192 * sizeof(float);
  float*  buf_at  = (float*)ws;  ws += N * 96 * sizeof(float);
  bf16_t* buf_h   = (bf16_t*)ws;                                  // N x 1024 bf16

  hipMemcpyAsync(buf_x, feat, N * DM * sizeof(float),
                 hipMemcpyDeviceToDevice, stream);

  const dim3 blk(256);
  const int MB = N_TOK / 128;  // 525 M-tiles

  for (int L = 0; L < 2; ++L) {
    // value / sampling-offset / attention-logit projections
    gemm_bias_kernel<0, false, false><<<dim3(MB, 4), blk, 0, stream>>>(
        buf_x, P(L, 4), P(L, 5), buf_v, 256, 256);
    gemm_bias_kernel<0, false, false><<<dim3(MB, 3), blk, 0, stream>>>(
        buf_x, P(L, 0), P(L, 1), buf_off, 256, 192);
    gemm_bias_kernel<0, false, false><<<dim3(MB, 2), blk, 0, stream>>>(
        buf_x, P(L, 2), P(L, 3), buf_at, 256, 96);

    // multi-scale deformable sampling (softmax + bilinear gather)
    deform_attn_kernel<<<dim3(N_TOK), blk, 0, stream>>>(buf_v, buf_off, buf_at,
                                                        buf_ao);

    // output projection, residual + LN1
    gemm_bias_kernel<0, false, false><<<dim3(MB, 4), blk, 0, stream>>>(
        buf_ao, P(L, 6), P(L, 7), buf_v, 256, 256);
    add_ln_kernel<<<dim3(QTOT), blk, 0, stream>>>(buf_x, buf_v, P(L, 12),
                                                  P(L, 13), buf_y);

    // FFN: GELU(y@W1+b1) (bf16 intermediate) then @W2+b2
    gemm_bias_kernel<1, false, true><<<dim3(MB, 16), blk, 0, stream>>>(
        buf_y, P(L, 8), P(L, 9), buf_h, 256, 1024);
    gemm_bias_kernel<0, true, false><<<dim3(MB, 4), blk, 0, stream>>>(
        buf_h, P(L, 10), P(L, 11), buf_ao, 1024, 256);

    // residual + LN2 -> next layer input (or final output)
    float* xout = (L == 1) ? (float*)d_out : buf_x;
    add_ln_kernel<<<dim3(QTOT), blk, 0, stream>>>(buf_y, buf_ao, P(L, 14),
                                                  P(L, 15), xout);
  }
}
